// EdgeConv_59648505807254
// MI455X (gfx1250) — matrix operand order repaired and checked by reading, not compiled, not run on hardware
//
#include <hip/hip_runtime.h>

// ---------------- problem constants (from reference) ----------------
#define BN   8
#define NN   4096
#define DD   16
#define OUTC 64
#define KK   20
#define EPSV 1e-5f

// ---------------- WMMA types (gfx1250, wave32) ----------------
typedef __attribute__((ext_vector_type(16))) __bf16 v16bf;
typedef __attribute__((ext_vector_type(8)))  float  v8f;

union ABreg { v16bf v; unsigned short u[16]; };
union CDreg { v8f   v; float e[8]; };

static __device__ __forceinline__ unsigned short f2bf(float f) {
  union { float f; unsigned u; } x; x.f = f;
  unsigned r = x.u + 0x7FFFu + ((x.u >> 16) & 1u);   // round-to-nearest-even
  return (unsigned short)(r >> 16);
}

static __device__ __forceinline__ void wave_fence() {
#if __has_builtin(__builtin_amdgcn_wave_barrier)
  __builtin_amdgcn_wave_barrier();
#else
  asm volatile("" ::: "memory");
#endif
}

// =====================================================================
// Kernel 1: fused pairwise-distance (bf16 WMMA, f32 accum) + top-k
// one block = one batch b, one tile of 16 query points. 256 threads.
// LDS holds the full 16 x 4096 d2 row-block (256 KB).
// =====================================================================
__global__ __launch_bounds__(256) void k_knn(const float* __restrict__ x,
                                             int* __restrict__ idxout)
{
  __shared__ float s_d2[16][NN];      // 256 KB
  __shared__ float s_sqq[16];

  const int tid  = threadIdx.x;
  const int wave = tid >> 5;
  const int lane = tid & 31;
  const int half = lane >> 4;
  const int col  = lane & 15;

  const int b  = blockIdx.x >> 8;          // NN/16 = 256 tiles per batch
  const int n0 = (blockIdx.x & 255) << 4;

  const float* xb = x + (size_t)b * NN * DD;

  // ---- A operand: 16x32 query tile (real K = 0..15, K 16..31 zero) ----
  // 16-bit A layout: lanes 0-15 -> M=lane, elems0-7 = K 0-7 ; lanes16-31 -> M=lane-16, elems0-7 = K 8-15
  ABreg a;
  float sqq_total;
  {
    const float4* qp = reinterpret_cast<const float4*>(xb + (size_t)(n0 + col) * DD + half * 8);
    float4 q0 = qp[0], q1 = qp[1];
    float ss = q0.x*q0.x + q0.y*q0.y + q0.z*q0.z + q0.w*q0.w
             + q1.x*q1.x + q1.y*q1.y + q1.z*q1.z + q1.w*q1.w;
    sqq_total = ss + __shfl_xor(ss, 16, 32);   // exact f32 |x_q|^2
    a.u[0]=f2bf(q0.x); a.u[1]=f2bf(q0.y); a.u[2]=f2bf(q0.z); a.u[3]=f2bf(q0.w);
    a.u[4]=f2bf(q1.x); a.u[5]=f2bf(q1.y); a.u[6]=f2bf(q1.z); a.u[7]=f2bf(q1.w);
#pragma unroll
    for (int j = 8; j < 16; ++j) a.u[j] = 0;
  }
  if (half == 0) s_sqq[col] = sqq_total;
  __syncthreads();

  // ---- phase B: each wave computes 16 x 512 of the d2 block ----
  for (int t = 0; t < 32; ++t) {
    const int m0 = wave * 512 + t * 16;

    // B operand: 32x16, column n = col holds neighbor m0+col.
    // lane half h: elems0-7 = K 8h..8h+7 (real), elems8-15 = K 16+8h.. (zero pad).
    ABreg bb;
#pragma unroll
    for (int j = 0; j < 16; ++j) bb.u[j] = 0;
    float sqm = 0.0f;
    {
      const float4* np = reinterpret_cast<const float4*>(xb + (size_t)(m0 + col) * DD);
      float4 v0 = np[0], v1 = np[1], v2 = np[2], v3 = np[3];
      sqm = v0.x*v0.x + v0.y*v0.y + v0.z*v0.z + v0.w*v0.w
          + v1.x*v1.x + v1.y*v1.y + v1.z*v1.z + v1.w*v1.w
          + v2.x*v2.x + v2.y*v2.y + v2.z*v2.z + v2.w*v2.w
          + v3.x*v3.x + v3.y*v3.y + v3.z*v3.z + v3.w*v3.w;
      const float4 p0 = half ? v2 : v0, p1 = half ? v3 : v1;
      bb.u[0] =f2bf(p0.x); bb.u[1] =f2bf(p0.y); bb.u[2] =f2bf(p0.z); bb.u[3] =f2bf(p0.w);
      bb.u[4] =f2bf(p1.x); bb.u[5] =f2bf(p1.y); bb.u[6] =f2bf(p1.z); bb.u[7] =f2bf(p1.w);
    }
    const float sqm_col = __shfl(sqm, col, 32);   // |x_m|^2 for this lane's column

    CDreg c;
#pragma unroll
    for (int j = 0; j < 8; ++j) c.e[j] = 0.0f;
    c.v = __builtin_amdgcn_wmma_f32_16x16x32_bf16(
        false, a.v, false, bb.v, (short)0, c.v, false, false);

    // D layout: vgpr r -> M = r + 8*half (query), N = col (neighbor in tile)
#pragma unroll
    for (int r = 0; r < 8; ++r) {
      const int q = r + 8 * half;
      s_d2[q][m0 + col] = s_sqq[q] + sqm_col - 2.0f * c.e[r];
    }
  }
  __syncthreads();

  // ---- phase C: top-k (k=20 smallest) per query; wave handles 2 queries ----
  for (int qi = 0; qi < 2; ++qi) {
    const int q = wave * 2 + qi;
    float* row = &s_d2[q][0];
    int* outp = idxout + ((size_t)b * NN + (n0 + q)) * KK;
    const float4* r4 = reinterpret_cast<const float4*>(row);

    for (int sel = 0; sel < KK; ++sel) {
      float bv = 3.0e38f; int bi = 0;
#pragma unroll 4
      for (int c4 = lane; c4 < NN / 4; c4 += 32) {
        float4 v = r4[c4];
        const int mb = c4 * 4;
        if (v.x < bv) { bv = v.x; bi = mb;     }
        if (v.y < bv) { bv = v.y; bi = mb + 1; }
        if (v.z < bv) { bv = v.z; bi = mb + 2; }
        if (v.w < bv) { bv = v.w; bi = mb + 3; }
      }
      // wave32 argmin reduction, tie-break on smallest index (matches top_k)
#pragma unroll
      for (int mask = 16; mask >= 1; mask >>= 1) {
        float ov = __shfl_xor(bv, mask, 32);
        int   oi = __shfl_xor(bi, mask, 32);
        if (ov < bv || (ov == bv && oi < bi)) { bv = ov; bi = oi; }
      }
      if (lane == 0) {
        outp[sel] = bi;
        row[bi] = 3.0e38f;        // knock out; DS ops are in-order per wave
      }
      wave_fence();
    }
  }
}

// =====================================================================
// Kernel 2: gather neighbors, build edge features, h = edge @ W^T
// one block = (b, 16-query tile). 8 waves; wave owns one 16-col slice of
// the 64 output channels and 10 of the 20 row tiles. bf16 WMMA 16x16x32.
// =====================================================================
__global__ __launch_bounds__(256) void k_edge(const float* __restrict__ x,
                                              const float* __restrict__ W,
                                              const int* __restrict__ idxin,
                                              float* __restrict__ out)
{
  __shared__ float s_xq[16][16];   // query rows
  __shared__ int   s_idx[16 * KK]; // 320 neighbor indices

  const int tid  = threadIdx.x;
  const int wave = tid >> 5;
  const int lane = tid & 31;
  const int half = lane >> 4;
  const int col  = lane & 15;

  const int b  = blockIdx.x >> 8;
  const int n0 = (blockIdx.x & 255) << 4;
  const float* xb = x + (size_t)b * NN * DD;

  // stage query tile + indices
  ((float*)s_xq)[tid] = xb[(size_t)n0 * DD + tid];
  {
    const int* ib = idxin + ((size_t)b * NN + n0) * KK;
    s_idx[tid] = ib[tid];
    if (tid < 64) s_idx[256 + tid] = ib[256 + tid];
  }
  __syncthreads();

  // ---- B operand: W^T tile for this wave's 16 output channels ----
  // B[k][n] = W[obase+n][k]; lane holds column n=col, K = 8*half + (0..7), 16 + 8*half + (0..7)
  const int nt    = wave & 3;
  const int obase = nt * 16;
  const int o     = obase + col;
  ABreg wb;
  {
    const float4* wp = reinterpret_cast<const float4*>(W + (size_t)o * (2 * DD) + half * 8);
    float4 w0 = wp[0], w1 = wp[1], w2 = wp[4], w3 = wp[5];
    wb.u[0] =f2bf(w0.x); wb.u[1] =f2bf(w0.y); wb.u[2] =f2bf(w0.z); wb.u[3] =f2bf(w0.w);
    wb.u[4] =f2bf(w1.x); wb.u[5] =f2bf(w1.y); wb.u[6] =f2bf(w1.z); wb.u[7] =f2bf(w1.w);
    wb.u[8] =f2bf(w2.x); wb.u[9] =f2bf(w2.y); wb.u[10]=f2bf(w2.z); wb.u[11]=f2bf(w2.w);
    wb.u[12]=f2bf(w3.x); wb.u[13]=f2bf(w3.y); wb.u[14]=f2bf(w3.z); wb.u[15]=f2bf(w3.w);
  }

  const int tbase = (wave >> 2) * 10;
  float* ob = out + ((size_t)b * OUTC + o) * ((size_t)NN * KK);

  for (int tt = 0; tt < 10; ++tt) {
    const int t = tbase + tt;

    // ---- A operand: 16 edge rows, K=0..31 = [center(16) | feature-center(16)]
    const int g  = t * 16 + col;       // row index = q*20 + kk
    const int q  = g / KK;
    const int xi = s_idx[g];

    float4 c0 = *reinterpret_cast<const float4*>(&s_xq[q][8 * half]);
    float4 c1 = *reinterpret_cast<const float4*>(&s_xq[q][8 * half + 4]);
    const float4* np = reinterpret_cast<const float4*>(xb + (size_t)xi * DD + 8 * half);
    float4 f0 = np[0], f1 = np[1];

    ABreg a;
    a.u[0] =f2bf(c0.x); a.u[1] =f2bf(c0.y); a.u[2] =f2bf(c0.z); a.u[3] =f2bf(c0.w);
    a.u[4] =f2bf(c1.x); a.u[5] =f2bf(c1.y); a.u[6] =f2bf(c1.z); a.u[7] =f2bf(c1.w);
    a.u[8] =f2bf(f0.x - c0.x); a.u[9] =f2bf(f0.y - c0.y);
    a.u[10]=f2bf(f0.z - c0.z); a.u[11]=f2bf(f0.w - c0.w);
    a.u[12]=f2bf(f1.x - c1.x); a.u[13]=f2bf(f1.y - c1.y);
    a.u[14]=f2bf(f1.z - c1.z); a.u[15]=f2bf(f1.w - c1.w);

    CDreg c;
#pragma unroll
    for (int j = 0; j < 8; ++j) c.e[j] = 0.0f;
    c.v = __builtin_amdgcn_wmma_f32_16x16x32_bf16(
        false, a.v, false, wb.v, (short)0, c.v, false, false);

    // scatter to (b, o, n, k) layout
#pragma unroll
    for (int r = 0; r < 8; ++r) {
      const int row = t * 16 + r + 8 * half;
      const int qq  = row / KK;
      const int kk  = row - qq * KK;
      ob[(size_t)(n0 + qq) * KK + kk] = c.e[r];
    }
  }
}

// =====================================================================
// Kernel 3: per-channel mean/var. one block per output channel.
// =====================================================================
__global__ __launch_bounds__(256) void k_stats(const float* __restrict__ h,
                                               float* __restrict__ stats)
{
  const int o   = blockIdx.x;
  const int tid = threadIdx.x;
  const size_t cs = (size_t)NN * KK;   // 81920 elems per (b, o)

  float s = 0.0f, s2 = 0.0f;
  for (int b = 0; b < BN; ++b) {
    const float4* p = reinterpret_cast<const float4*>(h + ((size_t)b * OUTC + o) * cs);
    for (int i = tid; i < (int)(cs / 4); i += 256) {
      float4 v = p[i];
      s  += v.x + v.y + v.z + v.w;
      s2 += v.x*v.x + v.y*v.y + v.z*v.z + v.w*v.w;
    }
  }
#pragma unroll
  for (int mask = 16; mask >= 1; mask >>= 1) {
    s  += __shfl_xor(s,  mask, 32);
    s2 += __shfl_xor(s2, mask, 32);
  }
  __shared__ float rs[8], rs2[8];
  const int wave = tid >> 5, lane = tid & 31;
  if (lane == 0) { rs[wave] = s; rs2[wave] = s2; }
  __syncthreads();
  if (tid == 0) {
    float S = 0.0f, S2 = 0.0f;
#pragma unroll
    for (int i = 0; i < 8; ++i) { S += rs[i]; S2 += rs2[i]; }
    const float inv  = 1.0f / (float)(BN * cs);
    const float mean = S * inv;
    const float var  = S2 * inv - mean * mean;
    stats[2 * o]     = mean;
    stats[2 * o + 1] = var;
  }
}

// =====================================================================
// Kernel 4: in-place batchnorm affine + ReLU (float4 streams)
// =====================================================================
__global__ __launch_bounds__(256) void k_norm(float* __restrict__ h,
                                              const float* __restrict__ stats,
                                              const float* __restrict__ gamma,
                                              const float* __restrict__ beta)
{
  const size_t i4 = (size_t)blockIdx.x * 256 + threadIdx.x;   // float4 index
  const size_t i  = i4 * 4;
  const int o = (int)((i / ((size_t)NN * KK)) % OUTC);

  const float mean = stats[2 * o];
  const float var  = stats[2 * o + 1];
  const float scl  = __frsqrt_rn(var + EPSV) * gamma[o];
  const float bia  = beta[o] - mean * scl;

  float4* p = reinterpret_cast<float4*>(h);
  float4 v = p[i4];
  v.x = fmaxf(v.x * scl + bia, 0.0f);
  v.y = fmaxf(v.y * scl + bia, 0.0f);
  v.z = fmaxf(v.z * scl + bia, 0.0f);
  v.w = fmaxf(v.w * scl + bia, 0.0f);
  p[i4] = v;
}

// =====================================================================
extern "C" void kernel_launch(void* const* d_in, const int* in_sizes, int n_in,
                              void* d_out, int out_size, void* d_ws, size_t ws_size,
                              hipStream_t stream)
{
  const float* x     = (const float*)d_in[0];
  const float* W     = (const float*)d_in[1];
  const float* gamma = (const float*)d_in[2];
  const float* beta  = (const float*)d_in[3];
  float* out = (float*)d_out;

  int*   idxbuf = (int*)d_ws;                                           // B*N*K ints (~2.6 MB)
  float* stats  = (float*)((char*)d_ws + (size_t)BN * NN * KK * sizeof(int)); // 128 floats

  const dim3 blk(256);
  k_knn  <<<BN * (NN / 16), blk, 0, stream>>>(x, idxbuf);
  k_edge <<<BN * (NN / 16), blk, 0, stream>>>(x, W, idxbuf, out);
  k_stats<<<OUTC,           blk, 0, stream>>>(out, stats);
  const size_t total = (size_t)BN * OUTC * NN * KK;                     // 41,943,040
  k_norm <<<(unsigned)(total / 1024), blk, 0, stream>>>(out, stats, gamma, beta);
}
